// MyGNN_47390669144219
// MI455X (gfx1250) — compile-verified, matrix operand order checked by reading, not run on hardware
//
#include <hip/hip_runtime.h>
#include <hip/hip_bf16.h>

typedef __attribute__((ext_vector_type(16))) _Float16 v16h;
typedef __attribute__((ext_vector_type(8)))  float    v8f;

#define N_NODES_C   50000
#define N_EDGES_C   800000
#define FD          64          // all feature dims are 64
#define MSG_CHUNKS  7           // packed K = 224 = 64(x_j)+64(x_i)+65(strain+eattr)+31(pad)
#define EDGE_CHUNKS 2           // K = 64
#define UPD_CHUNKS  4           // K = 128
#define LSTRIDE     72          // LDS row stride (floats)

#define MSG_FRAG_ELEMS  (MSG_CHUNKS * 4 * 32 * 16)   // 14336 halves
#define EDGE_FRAG_ELEMS (EDGE_CHUNKS * 4 * 32 * 16)  //  4096 halves
#define UPD_FRAG_ELEMS  (UPD_CHUNKS * 4 * 32 * 16)   //  8192 halves

union HV { v16h v; _Float16 h[16]; };

// ---------------------------------------------------------------------------
// Weight pre-pack: f32 [K,64] -> f16 WMMA B-fragments.
// Fragment element (chunk c, ntile t, lane l, half j) holds
//   W[ 32*c + (j&7) + 16*(j>>3) + 8*(l>>4) ][ 16*t + (l&15) ]
// (rows >= kMax are zero padding).
// ---------------------------------------------------------------------------
__global__ __launch_bounds__(256) void pack_weights_k(
    const float* __restrict__ Wmsg, const float* __restrict__ Wedge,
    const float* __restrict__ Wupd,
    _Float16* __restrict__ pMsg, _Float16* __restrict__ pEdge,
    _Float16* __restrict__ pUpd)
{
    int tid = blockIdx.x * 256 + threadIdx.x;
    const float* W; _Float16* dst; int kMax; int idx;
    if (tid < MSG_FRAG_ELEMS) {
        W = Wmsg; dst = pMsg; kMax = 193; idx = tid;
    } else if (tid < MSG_FRAG_ELEMS + EDGE_FRAG_ELEMS) {
        W = Wedge; dst = pEdge; kMax = 64; idx = tid - MSG_FRAG_ELEMS;
    } else if (tid < MSG_FRAG_ELEMS + EDGE_FRAG_ELEMS + UPD_FRAG_ELEMS) {
        W = Wupd; dst = pUpd; kMax = 128; idx = tid - MSG_FRAG_ELEMS - EDGE_FRAG_ELEMS;
    } else {
        return;
    }
    int j    = idx & 15;
    int lane = (idx >> 4) & 31;
    int ct   = idx >> 9;
    int t    = ct & 3;
    int c    = ct >> 2;
    int k = (j & 7) + 16 * (j >> 3) + 8 * (lane >> 4) + 32 * c;
    int n = t * 16 + (lane & 15);
    float v = (k < kMax) ? W[(size_t)k * FD + n] : 0.0f;
    dst[idx] = (_Float16)v;
}

__device__ __forceinline__ v16h load_bfrag(const _Float16* __restrict__ p,
                                           int c, int t, int lane) {
    return *(const v16h*)(p + (size_t)(((c * 4 + t) * 32 + lane) << 4));
}

// Convert 8+8 contiguous f32 (K..K+7, K+16..K+23) into an A-fragment lane slice.
__device__ __forceinline__ v16h cvt_afrag(float4 f0, float4 f1, float4 f2, float4 f3,
                                          float s) {
    HV u;
    u.h[0]  = (_Float16)(f0.x * s); u.h[1]  = (_Float16)(f0.y * s);
    u.h[2]  = (_Float16)(f0.z * s); u.h[3]  = (_Float16)(f0.w * s);
    u.h[4]  = (_Float16)(f1.x * s); u.h[5]  = (_Float16)(f1.y * s);
    u.h[6]  = (_Float16)(f1.z * s); u.h[7]  = (_Float16)(f1.w * s);
    u.h[8]  = (_Float16)(f2.x * s); u.h[9]  = (_Float16)(f2.y * s);
    u.h[10] = (_Float16)(f2.z * s); u.h[11] = (_Float16)(f2.w * s);
    u.h[12] = (_Float16)(f3.x * s); u.h[13] = (_Float16)(f3.y * s);
    u.h[14] = (_Float16)(f3.z * s); u.h[15] = (_Float16)(f3.w * s);
    return u.v;
}

// ---------------------------------------------------------------------------
// Edge kernel: per wave, one 16-edge tile.
//   msg   = softplus([x_j | x_i | strain | e_attr] @ Wmsg + b)    (7x4 WMMA)
//   scatter-add msg into agg (= x_new region of d_out) + counts
//   edge_new = msg @ Wedge + b                                    (2x4 WMMA)
// ---------------------------------------------------------------------------
__global__ __launch_bounds__(256) void edge_kernel(
    const float* __restrict__ x, const int* __restrict__ eidx,
    const float* __restrict__ eattr, const float* __restrict__ r,
    const float* __restrict__ dinit,
    const _Float16* __restrict__ pMsg, const _Float16* __restrict__ pEdge,
    const float* __restrict__ bMsg, const float* __restrict__ bEdge,
    float* __restrict__ aggOut, float* __restrict__ cnt,
    float* __restrict__ msgOut, float* __restrict__ edgeOut,
    float* __restrict__ strainOut)
{
    __shared__ float lds[8][16 * LSTRIDE];
    const int wave = threadIdx.x >> 5;
    const int lane = threadIdx.x & 31;
    const int tile = blockIdx.x * 8 + wave;     // N_EDGES/16 = 50000 tiles exact
    const int base = tile * 16;
    float* L = lds[wave];

    const int* __restrict__ srcIdx = eidx;
    const int* __restrict__ dstIdx = eidx + N_EDGES_C;

    const int m     = lane & 15;    // edge row within tile / N column within ntile
    const int hb    = lane >> 4;    // half-wave select
    const int kbase = hb * 8;

    // ---- stage [strain | edge_attr] into LDS (feat index 0 = strain, 1..64 = eattr)
#pragma unroll
    for (int i = 0; i < 8; ++i) {
        int flat = i * 32 + lane;              // 256 float4 slots for the tile
        int em = flat >> 4, q = flat & 15;
        float4 v = *(const float4*)(eattr + (size_t)(base + em) * FD + q * 4);
        float* p = &L[em * LSTRIDE + 1 + q * 4];
        p[0] = v.x; p[1] = v.y; p[2] = v.z; p[3] = v.w;
    }
    if (lane < 16) {
        int e = base + lane;
        float2 rv = ((const float2*)r)[e];
        float d  = sqrtf(rv.x * rv.x + rv.y * rv.y);
        float di = dinit[e];
        float s  = (d - di) / di;
        L[lane * LSTRIDE] = s;
        strainOut[e] = s;
    }
    __syncthreads();

    const int srcN = srcIdx[base + m];
    const int dstN = dstIdx[base + m];

    v8f acc[4];
#pragma unroll
    for (int t = 0; t < 4; ++t) {
        float bv = bMsg[t * 16 + m];
        acc[t] = (v8f){bv, bv, bv, bv, bv, bv, bv, bv};
    }

    // ---- chunks 0..3: gathered node features (x_j: 0-1, x_i: 2-3)
#pragma unroll
    for (int c = 0; c < 4; ++c) {
        const float* row = x + (size_t)((c < 2) ? srcN : dstN) * FD + (c & 1) * 32 + kbase;
        float4 f0 = *(const float4*)(row);
        float4 f1 = *(const float4*)(row + 4);
        float4 f2 = *(const float4*)(row + 16);
        float4 f3 = *(const float4*)(row + 20);
        v16h a = cvt_afrag(f0, f1, f2, f3, 1.0f);
#pragma unroll
        for (int t = 0; t < 4; ++t) {
            v16h b = load_bfrag(pMsg, c, t, lane);
            acc[t] = __builtin_amdgcn_wmma_f32_16x16x32_f16(
                false, a, false, b, (short)0, acc[t], false, false);
        }
    }
    // ---- chunks 4..6: strain + edge_attr from LDS (feat idx = packedRow - 128)
#pragma unroll
    for (int c = 4; c < 7; ++c) {
        int ib = (c - 4) * 32 + kbase;
        HV u;
#pragma unroll
        for (int j = 0; j < 16; ++j) {
            int idx = ib + ((j < 8) ? j : j + 8);
            float v = (idx <= 64) ? L[m * LSTRIDE + idx] : 0.0f;
            u.h[j] = (_Float16)v;
        }
#pragma unroll
        for (int t = 0; t < 4; ++t) {
            v16h b = load_bfrag(pMsg, c, t, lane);
            acc[t] = __builtin_amdgcn_wmma_f32_16x16x32_f16(
                false, u.v, false, b, (short)0, acc[t], false, false);
        }
    }

    // ---- softplus (stable): max(x,0) + log(1 + exp(-|x|))
#pragma unroll
    for (int t = 0; t < 4; ++t)
#pragma unroll
        for (int v = 0; v < 8; ++v) {
            float xv = acc[t][v];
            acc[t][v] = fmaxf(xv, 0.0f) + __logf(1.0f + __expf(-fabsf(xv)));
        }

    __syncthreads();   // done reading feat LDS; reuse buffer for msg staging

    int dstRows[8];
#pragma unroll
    for (int v = 0; v < 8; ++v) dstRows[v] = dstIdx[base + v + 8 * hb];

    // ---- write msg, stage into LDS, scatter-add into agg
#pragma unroll
    for (int t = 0; t < 4; ++t) {
        int ncol = t * 16 + m;
#pragma unroll
        for (int v = 0; v < 8; ++v) {
            int row = v + 8 * hb;
            float val = acc[t][v];
            L[row * LSTRIDE + ncol] = val;
            msgOut[(size_t)(base + row) * FD + ncol] = val;
            atomicAdd(&aggOut[(size_t)dstRows[v] * FD + ncol], val);
        }
    }
    if (lane < 16) atomicAdd(&cnt[dstIdx[base + lane]], 1.0f);
    __syncthreads();

    // ---- edge_new = msg @ Wedge + b  (msg re-read from LDS in A-fragment layout)
    v8f eacc[4];
#pragma unroll
    for (int t = 0; t < 4; ++t) {
        float bv = bEdge[t * 16 + m];
        eacc[t] = (v8f){bv, bv, bv, bv, bv, bv, bv, bv};
    }
#pragma unroll
    for (int c = 0; c < 2; ++c) {
        HV u;
#pragma unroll
        for (int j = 0; j < 16; ++j) {
            int kk = c * 32 + kbase + ((j < 8) ? j : j + 8);
            u.h[j] = (_Float16)L[m * LSTRIDE + kk];
        }
#pragma unroll
        for (int t = 0; t < 4; ++t) {
            v16h b = load_bfrag(pEdge, c, t, lane);
            eacc[t] = __builtin_amdgcn_wmma_f32_16x16x32_f16(
                false, u.v, false, b, (short)0, eacc[t], false, false);
        }
    }
#pragma unroll
    for (int t = 0; t < 4; ++t)
#pragma unroll
        for (int v = 0; v < 8; ++v)
            edgeOut[(size_t)(base + v + 8 * hb) * FD + t * 16 + m] = eacc[t][v];
}

// ---------------------------------------------------------------------------
// Node kernel: x_new = [x | agg/max(cnt,1)] @ Wupd + b   (4x4 WMMA per tile).
// agg aliases the x_new output region; each wave reads only the 16 rows it
// subsequently overwrites.
// ---------------------------------------------------------------------------
__global__ __launch_bounds__(256) void node_kernel(
    const float* __restrict__ x, const float* __restrict__ agg,
    const float* __restrict__ cnt, const _Float16* __restrict__ pUpd,
    const float* __restrict__ bUpd, float* __restrict__ xnewOut)
{
    const int wave = threadIdx.x >> 5;
    const int lane = threadIdx.x & 31;
    const int tile = blockIdx.x * 8 + wave;
    if (tile >= N_NODES_C / 16) return;        // wave-uniform guard
    const int base  = tile * 16;
    const int m     = lane & 15;
    const int hb    = lane >> 4;
    const int kbase = hb * 8;
    const int node  = base + m;

    float rc = 1.0f / fmaxf(cnt[node], 1.0f);

    v8f acc[4];
#pragma unroll
    for (int t = 0; t < 4; ++t) {
        float bv = bUpd[t * 16 + m];
        acc[t] = (v8f){bv, bv, bv, bv, bv, bv, bv, bv};
    }
#pragma unroll
    for (int c = 0; c < 4; ++c) {
        const float* row;
        float s;
        if (c < 2) { row = x   + (size_t)node * FD + c * 32 + kbase;       s = 1.0f; }
        else       { row = agg + (size_t)node * FD + (c - 2) * 32 + kbase; s = rc;   }
        float4 f0 = *(const float4*)(row);
        float4 f1 = *(const float4*)(row + 4);
        float4 f2 = *(const float4*)(row + 16);
        float4 f3 = *(const float4*)(row + 20);
        v16h a = cvt_afrag(f0, f1, f2, f3, s);
#pragma unroll
        for (int t = 0; t < 4; ++t) {
            v16h b = load_bfrag(pUpd, c, t, lane);
            acc[t] = __builtin_amdgcn_wmma_f32_16x16x32_f16(
                false, a, false, b, (short)0, acc[t], false, false);
        }
    }
#pragma unroll
    for (int t = 0; t < 4; ++t)
#pragma unroll
        for (int v = 0; v < 8; ++v)
            xnewOut[(size_t)(base + v + 8 * hb) * FD + t * 16 + m] = acc[t][v];
}

// ---------------------------------------------------------------------------
extern "C" void kernel_launch(void* const* d_in, const int* in_sizes, int n_in,
                              void* d_out, int out_size, void* d_ws, size_t ws_size,
                              hipStream_t stream) {
    const float* x     = (const float*)d_in[0];
    const int*   eidx  = (const int*)d_in[1];
    const float* eattr = (const float*)d_in[2];
    const float* r     = (const float*)d_in[3];
    const float* dinit = (const float*)d_in[4];
    const float* Wmsg  = (const float*)d_in[5];
    const float* bMsg  = (const float*)d_in[6];
    const float* Wupd  = (const float*)d_in[7];
    const float* bUpd  = (const float*)d_in[8];
    const float* Wedge = (const float*)d_in[9];
    const float* bEdge = (const float*)d_in[10];

    float* out       = (float*)d_out;
    float* xnew      = out;                                         // 50000*64
    float* edgeNew   = out + (size_t)N_NODES_C * FD;                // 800000*64
    float* msgOut    = edgeNew + (size_t)N_EDGES_C * FD;            // 800000*64
    float* strainOut = msgOut + (size_t)N_EDGES_C * FD;             // 800000

    char* ws = (char*)d_ws;
    _Float16* pMsg  = (_Float16*)ws;                                      // 28672 B
    _Float16* pEdge = (_Float16*)(ws + MSG_FRAG_ELEMS * 2);               //  8192 B
    _Float16* pUpd  = (_Float16*)(ws + (MSG_FRAG_ELEMS + EDGE_FRAG_ELEMS) * 2); // 16384 B
    float* cnt = (float*)(ws + (MSG_FRAG_ELEMS + EDGE_FRAG_ELEMS + UPD_FRAG_ELEMS) * 2);

    // zero the agg accumulator (aliases x_new output) and the counts
    hipMemsetAsync(xnew, 0, (size_t)N_NODES_C * FD * sizeof(float), stream);
    hipMemsetAsync(cnt, 0, (size_t)N_NODES_C * sizeof(float), stream);

    pack_weights_k<<<(MSG_FRAG_ELEMS + EDGE_FRAG_ELEMS + UPD_FRAG_ELEMS) / 256, 256, 0,
                     stream>>>(Wmsg, Wedge, Wupd, pMsg, pEdge, pUpd);

    edge_kernel<<<(N_EDGES_C / 16) / 8, 256, 0, stream>>>(
        x, eidx, eattr, r, dinit, pMsg, pEdge, bMsg, bEdge,
        xnew /*agg*/, cnt, msgOut, edgeNew, strainOut);

    int nodeTiles = N_NODES_C / 16;
    node_kernel<<<(nodeTiles + 7) / 8, 256, 0, stream>>>(
        x, xnew /*agg*/, cnt, pUpd, bUpd, xnew);
}
